// S4Block_29824252903857
// MI455X (gfx1250) — matrix operand order repaired and checked by reading, not compiled, not run on hardware
//
#include <hip/hip_runtime.h>

#define BATCH 8
#define LEN   4096
#define CH    256
#define HID   32
#define L2X   8192
#define PI_F  3.14159265358979323846f

typedef __attribute__((ext_vector_type(16))) __bf16 bf16x16;
typedef __attribute__((ext_vector_type(8)))  float  floatx8;

#if __has_builtin(__builtin_amdgcn_global_load_async_to_lds_b128)
#define HAVE_ASYNC_LDS 1
#else
#define HAVE_ASYNC_LDS 0
#endif

// int4 vector + address-space-qualified flavors for the async-LDS builtin
typedef int v4i __attribute__((vector_size(16)));
typedef __attribute__((address_space(1))) v4i gv4i;   // global
typedef __attribute__((address_space(3))) v4i lv4i;   // LDS

__device__ __forceinline__ void wait_asynccnt0() {
#if __has_builtin(__builtin_amdgcn_s_wait_asynccnt)
    __builtin_amdgcn_s_wait_asynccnt(0);
#else
    asm volatile("s_wait_asynccnt 0x0" ::: "memory");
#endif
}

// ---------------- complex helpers ----------------
__device__ __forceinline__ float2 cmulf(float2 a, float2 b) {
    return make_float2(a.x * b.x - a.y * b.y, a.x * b.y + a.y * b.x);
}
__device__ __forceinline__ float2 cinvf(float2 a) {
    float d = a.x * a.x + a.y * a.y;
    float r = 1.0f / d;
    return make_float2(a.x * r, -a.y * r);
}
__device__ __forceinline__ unsigned short f2bf(float f) {
    unsigned int u = __float_as_uint(f);
    unsigned int r = u + 0x7FFFu + ((u >> 16) & 1u);   // round-to-nearest-even
    return (unsigned short)(r >> 16);
}

// ---------------- in-place radix-2 DIT FFT in LDS ----------------
__device__ void fft_lds(float2* buf, int N, int logN, float sign) {
    const int tid = threadIdx.x, T = blockDim.x;
    __syncthreads();
    for (int i = tid; i < N; i += T) {
        int j = (int)(__brev((unsigned)i) >> (32 - logN));
        if (j > i) { float2 t = buf[i]; buf[i] = buf[j]; buf[j] = t; }
    }
    __syncthreads();
    for (int hs = 0; hs < logN; ++hs) {
        const int half = 1 << hs;
        const float base = sign * (2.0f * PI_F) / (float)(half << 1);
        for (int idx = tid; idx < (N >> 1); idx += T) {
            const int off = idx & (half - 1);
            const int p = ((idx >> hs) << (hs + 1)) + off;
            float sn, cs;
            __sincosf(base * (float)off, &sn, &cs);
            float2 u = buf[p];
            float2 v = buf[p + half];
            float2 t = make_float2(v.x * cs - v.y * sn, v.x * sn + v.y * cs);
            buf[p]        = make_float2(u.x + t.x, u.y + t.y);
            buf[p + half] = make_float2(u.x - t.x, u.y - t.y);
        }
        __syncthreads();
    }
}

// ---------------- 1) LayerNorm ----------------
__global__ void ln_kernel(const float* __restrict__ x, const float* __restrict__ g,
                          const float* __restrict__ be, float* __restrict__ y) {
    __shared__ float red[CH];
    const int row = blockIdx.x, c = threadIdx.x;
    const float v = x[(size_t)row * CH + c];
    red[c] = v; __syncthreads();
    for (int s = CH / 2; s > 0; s >>= 1) { if (c < s) red[c] += red[c + s]; __syncthreads(); }
    const float mu = red[0] * (1.0f / CH);
    __syncthreads();
    const float d = v - mu;
    red[c] = d * d; __syncthreads();
    for (int s = CH / 2; s > 0; s >>= 1) { if (c < s) red[c] += red[c + s]; __syncthreads(); }
    const float var = red[0] * (1.0f / CH);
    y[(size_t)row * CH + c] = d * rsqrtf(var + 1e-5f) * g[c] + be[c];
}

// ---------------- 2) W -> bf16 ----------------
__global__ void w2bf_kernel(const float* __restrict__ W, unsigned short* __restrict__ wbf, int n) {
    int i = blockIdx.x * blockDim.x + threadIdx.x;
    if (i < n) wbf[i] = f2bf(W[i]);
}

// ---------------- 3) per-channel Cauchy -> iFFT(4096) -> FFT(8192) = Kf ----------------
__global__ void build_kf_kernel(const float* __restrict__ A_real, const float* __restrict__ A_imag,
                                const float* __restrict__ P_re,  const float* __restrict__ P_im,
                                const float* __restrict__ B_re,  const float* __restrict__ B_im,
                                const float* __restrict__ C_re,  const float* __restrict__ C_im,
                                const float* __restrict__ step,  float2* __restrict__ Kf) {
    extern __shared__ char smem[];
    float2* buf = (float2*)smem;          // 8192 complex
    float2* lam = buf + L2X;              // 32
    float2* p00 = lam + HID;
    float2* p01 = p00 + HID;
    float2* p10 = p01 + HID;
    float2* p11 = p10 + HID;

    const int c = blockIdx.x, tid = threadIdx.x, T = blockDim.x;
    if (tid < HID) {
        const int h = tid;
        lam[h] = make_float2(fminf(A_real[h], -1e-4f), A_imag[h]);
        float2 a0 = make_float2(C_re[c * HID + h], -C_im[c * HID + h]); // conj(Cs)
        float2 b0 = make_float2(B_re[h],  B_im[h]);
        float2 a1 = make_float2(P_re[h], -P_im[h]);                    // conj(P)
        float2 b1 = make_float2(P_re[h],  P_im[h]);
        p00[h] = cmulf(a0, b0); p01[h] = cmulf(a0, b1);
        p10[h] = cmulf(a1, b0); p11[h] = cmulf(a1, b1);
    }
    __syncthreads();

    const float st = step[c];
    const float dt = (st > 20.0f) ? st : log1pf(__expf(st));          // softplus
    const float two_dt = 2.0f / dt;

    for (int j = tid; j < LEN; j += T) {
        float sw, cw;
        __sincosf(-2.0f * PI_F * (float)j / (float)LEN, &sw, &cw);
        float2 onePw = make_float2(1.0f + cw,  sw);
        float2 oneMw = make_float2(1.0f - cw, -sw);
        float2 ip = cinvf(onePw);
        float2 gg = cmulf(oneMw, ip);
        gg.x *= two_dt; gg.y *= two_dt;
        float2 cf = make_float2(2.0f * ip.x, 2.0f * ip.y);

        float2 k00 = {0, 0}, k01 = {0, 0}, k10 = {0, 0}, k11 = {0, 0};
#pragma unroll 8
        for (int h = 0; h < HID; ++h) {
            float2 den = make_float2(gg.x - lam[h].x, gg.y - lam[h].y);
            float2 id  = cinvf(den);
            float2 t;
            t = cmulf(p00[h], id); k00.x += t.x; k00.y += t.y;
            t = cmulf(p01[h], id); k01.x += t.x; k01.y += t.y;
            t = cmulf(p10[h], id); k10.x += t.x; k10.y += t.y;
            t = cmulf(p11[h], id); k11.x += t.x; k11.y += t.y;
        }
        float2 d11 = make_float2(1.0f + k11.x, k11.y);
        float2 corr = cmulf(k01, cmulf(k10, cinvf(d11)));
        float2 at = make_float2(k00.x - corr.x, k00.y - corr.y);
        buf[j] = cmulf(cf, at);
    }

    fft_lds(buf, LEN, 12, +1.0f);                 // inverse DFT (unscaled)
    for (int t = tid; t < LEN; t += T) {
        buf[t].x *= (1.0f / (float)LEN);
        buf[t].y = 0.0f;
        buf[t + LEN] = make_float2(0.0f, 0.0f);
    }
    fft_lds(buf, L2X, 13, -1.0f);                 // forward: Kf spectrum

    float2* Kfc = Kf + (size_t)c * L2X;
    for (int j = tid; j < L2X; j += T) Kfc[j] = buf[j];
}

// ------- 4) per (batch,channel): async-stage Kf into LDS under the FFT, conv + D*y -------
__global__ void fftconv_kernel(const float* __restrict__ y, const float2* __restrict__ Kf,
                               const float* __restrict__ Dv, unsigned short* __restrict__ s16) {
    extern __shared__ char smem[];
    float2* buf = (float2*)smem;                  // 8192 complex (64 KB)
    float*  yl  = (float*)(buf + L2X);            // 4096 floats (16 KB)
    float2* kfl = (float2*)(yl + LEN);            // 8192 complex (64 KB) staged Kf

    const int bc = blockIdx.x;
    const int b = bc >> 8, c = bc & 255;
    const int tid = threadIdx.x, T = blockDim.x;

    const float2* Kfc = Kf + (size_t)c * L2X;
#if HAVE_ASYNC_LDS
    // Kick off the 64 KB Kf transfer now; it completes in the background while
    // the 13-stage forward FFT runs (ASYNCcnt-tracked, no VGPR staging).
    for (int j = tid; j < (L2X / 2); j += T) {    // 16 B per lane per op
        __builtin_amdgcn_global_load_async_to_lds_b128(
            (gv4i*)(Kfc + 2 * j),
            (lv4i*)(kfl + 2 * j), 0, 0);
    }
#endif

    for (int t = tid; t < LEN; t += T) {
        float yv = y[((size_t)(b * LEN + t)) * CH + c];
        yl[t] = yv;
        buf[t] = make_float2(yv, 0.0f);
        buf[t + LEN] = make_float2(0.0f, 0.0f);
    }
    fft_lds(buf, L2X, 13, -1.0f);                 // Y(f)

#if HAVE_ASYNC_LDS
    wait_asynccnt0();                             // our wave's transfers done
    __syncthreads();                              // everyone's transfers visible
    for (int j = tid; j < L2X; j += T) buf[j] = cmulf(buf[j], kfl[j]);
#else
    for (int j = tid; j < L2X; j += T) buf[j] = cmulf(buf[j], Kfc[j]);
#endif

    fft_lds(buf, L2X, 13, +1.0f);                 // inverse (unscaled)

    const float Dc = Dv[c];
    for (int t = tid; t < LEN; t += T) {
        float sv = buf[t].x * (1.0f / (float)L2X) + Dc * yl[t];
        s16[((size_t)(b * LEN + t)) * CH + c] = f2bf(sv);
    }
}

// ------- 5) out = gelu(s @ W^T + b) + x : 16x32 tile/wave, v_wmma_f32_16x16x32_bf16 -------
__global__ void __launch_bounds__(128)
gemm_gelu_kernel(const unsigned short* __restrict__ s16, const unsigned short* __restrict__ wbf,
                 const float* __restrict__ bias, const float* __restrict__ x,
                 float* __restrict__ out) {
    const int lane = threadIdx.x & 31;
    const int wid  = threadIdx.x >> 5;
    const int g    = blockIdx.x * 4 + wid;        // tile-pair id, 0..16383
    const int tn   = (g & 7) << 1;                // even N tile (pairs of 16)
    const int tm   = g >> 3;                      // M tile, 0..2047
    const int m0   = tm << 4, n0 = tn << 4;

    const int rowm = lane & 15;                   // A M-row / B N-column of this lane
    const int hi   = lane >> 4;
    const int kb2A = hi << 3;                     // A u32 sub-offset (0 or 8)
    const int kb2B = hi << 3;                     // B u32 sub-offset (0 or 8)

    const unsigned int* ap  = (const unsigned int*)(s16 + (size_t)(m0 + rowm) * CH);
    const unsigned int* bp0 = (const unsigned int*)(wbf + (size_t)(n0 + rowm) * CH);
    const unsigned int* bp1 = (const unsigned int*)(wbf + (size_t)(n0 + 16 + rowm) * CH);

    floatx8 acc0 = {}, acc1 = {};
    union Frag { bf16x16 v; unsigned int u[8]; };
#pragma unroll
    for (int kk = 0; kk < 8; ++kk) {              // K = 256, 32 per WMMA
        const int base = kk << 4;
        if (kk < 7) {
            __builtin_prefetch(ap + base + 16, 0, 1);    // -> global_prefetch_b8
            __builtin_prefetch(bp0 + base + 16, 0, 1);
        }
        Frag A, B0, B1;
#pragma unroll
        for (int v = 0; v < 4; ++v) {
            A.u[v]     = ap[base + kb2A + v];     // K = kb + 0..7
            A.u[4 + v] = ap[base + 4 + kb2A + v]; // K = kb + 8..15
        }
#pragma unroll
        for (int v = 0; v < 8; ++v) {
            B0.u[v] = bp0[base + kb2B + v];       // K = kbB + 0..15
            B1.u[v] = bp1[base + kb2B + v];
        }
        acc0 = __builtin_amdgcn_wmma_f32_16x16x32_bf16(
            false, A.v, false, B0.v, (short)0, acc0, false, false);
        acc1 = __builtin_amdgcn_wmma_f32_16x16x32_bf16(
            false, A.v, false, B1.v, (short)0, acc1, false, false);
    }

    // C/D layout: VGPR r -> (M = r + 8*hi, N = lane&15)
#pragma unroll
    for (int r = 0; r < 8; ++r) {
        const int m = m0 + r + (hi << 3);
        const int n = n0 + rowm;
        float v0 = acc0[r] + bias[n];
        float v1 = acc1[r] + bias[n + 16];
        float g0 = 0.5f * v0 * (1.0f + erff(v0 * 0.70710678118654752f));
        float g1 = 0.5f * v1 * (1.0f + erff(v1 * 0.70710678118654752f));
        out[(size_t)m * CH + n]      = g0 + x[(size_t)m * CH + n];
        out[(size_t)m * CH + n + 16] = g1 + x[(size_t)m * CH + n + 16];
    }
}

// ---------------- launcher ----------------
extern "C" void kernel_launch(void* const* d_in, const int* in_sizes, int n_in,
                              void* d_out, int out_size, void* d_ws, size_t ws_size,
                              hipStream_t stream) {
    (void)in_sizes; (void)n_in; (void)out_size; (void)ws_size;
    const float* x     = (const float*)d_in[0];
    const float* gam   = (const float*)d_in[1];
    const float* bet   = (const float*)d_in[2];
    const float* Areal = (const float*)d_in[3];
    const float* Aimag = (const float*)d_in[4];
    const float* Pre   = (const float*)d_in[5];
    const float* Pim   = (const float*)d_in[6];
    const float* Bre   = (const float*)d_in[7];
    const float* Bim   = (const float*)d_in[8];
    const float* Cre   = (const float*)d_in[9];
    const float* Cim   = (const float*)d_in[10];
    const float* Dv    = (const float*)d_in[11];
    const float* stepv = (const float*)d_in[12];
    const float* W     = (const float*)d_in[13];
    const float* bias  = (const float*)d_in[14];

    char* ws = (char*)d_ws;
    float*          y   = (float*)(ws);                            // 33,554,432 B
    float2*         Kf  = (float2*)(ws + 33554432);                // 16,777,216 B
    unsigned short* s16 = (unsigned short*)(ws + 50331648);        // 16,777,216 B
    unsigned short* wbf = (unsigned short*)(ws + 67108864);        //    131,072 B

    ln_kernel<<<BATCH * LEN, CH, 0, stream>>>(x, gam, bet, y);
    w2bf_kernel<<<(CH * CH + 255) / 256, 256, 0, stream>>>(W, wbf, CH * CH);

    const size_t smemB = (size_t)L2X * sizeof(float2) + (size_t)(5 * HID) * sizeof(float2);
    build_kf_kernel<<<CH, 256, smemB, stream>>>(Areal, Aimag, Pre, Pim, Bre, Bim,
                                                Cre, Cim, stepv, Kf);

    const size_t smemC = (size_t)L2X * sizeof(float2) + (size_t)LEN * sizeof(float)
                       + (size_t)L2X * sizeof(float2);             // buf + y + staged Kf
    fftconv_kernel<<<BATCH * CH, 256, smemC, stream>>>(y, Kf, Dv, s16);

    gemm_gelu_kernel<<<(BATCH * LEN / 16) * (CH / 32) / 4, 128, 0, stream>>>(
        s16, wbf, bias, x, (float*)d_out);
}
